// MultiHeadAttention_11519102288232
// MI455X (gfx1250) — matrix operand order repaired and checked by reading, not compiled, not run on hardware
//
#include <hip/hip_runtime.h>
#include <hip/hip_bf16.h>

typedef __attribute__((ext_vector_type(16))) __bf16 v16bf;
typedef __attribute__((ext_vector_type(8)))  float  v8f;

union FragBF { v16bf v; unsigned u[8]; };

#define HID   1024
#define HEADS 16
#define HDIM  64
#define SEQ   2048
#define BATCH 2

// ---- gfx1250 async global->LDS copy (ASYNCcnt path), 16B granule ----
__device__ __forceinline__ void async_copy_b128(const void* gaddr, unsigned lds_off) {
    asm volatile("global_load_async_to_lds_b128 %0, %1, off"
                 :: "v"(lds_off), "v"(gaddr) : "memory");
}
__device__ __forceinline__ void wait_async_0() {
    asm volatile("s_wait_asynccnt 0x0" ::: "memory");
}
__device__ __forceinline__ unsigned lds_off_of(const void* p) {
    // generic pointers to __shared__ carry the wave-relative LDS offset in addr[31:0]
    return (unsigned)(size_t)p;
}

// ---------------- f32 -> bf16 conversion ----------------
__global__ void cvt_f32_to_bf16(const float* __restrict__ in, __bf16* __restrict__ out, int n) {
    int i = blockIdx.x * blockDim.x + threadIdx.x;
    if (i < n) out[i] = (__bf16)in[i];
}

// ---------------- bf16 WMMA GEMM: C = A[M,K] * B[K,N] + bias ----------------
// mode 0: store bf16 split-head [B,HEADS,SEQ,HDIM]
// mode 1: store f32 transposed  [B,HID,SEQ]
__global__ __launch_bounds__(128) void gemm_bf16_wmma(
    const __bf16* __restrict__ A,
    const __bf16* __restrict__ Bm,
    const float*  __restrict__ bias,
    void* __restrict__ outp,
    int M, int N, int K, int mode)
{
    __shared__ __bf16 As[64][40];   // 80B rows: 16B-aligned for async b128, conflict-free frag reads
    __shared__ __bf16 Bs[64][40];   // [n][k] transposed on fill

    const int tid  = threadIdx.x;
    const int lane = tid & 31;
    const int wave = tid >> 5;
    const int m0 = blockIdx.x * 64;
    const int n0 = blockIdx.y * 64;
    const int wm = (wave & 1) * 32;
    const int wn = (wave >> 1) * 32;

    const int halo = lane >> 4;
    const int l16  = lane & 15;
    const int kbA  = halo * 8;
    const int kbB  = halo * 16;

    v8f acc[2][2] = {};

    for (int k0 = 0; k0 < K; k0 += 32) {
        // A tile 64x32 via async b128: 256 chunks of 16B, 2 per thread
        #pragma unroll
        for (int i = 0; i < 2; ++i) {
            int c   = i * 128 + tid;
            int row = c >> 2;            // 4 chunks per 32-elem row
            int j   = c & 3;
            async_copy_b128(A + (size_t)(m0 + row) * K + k0 + 8 * j,
                            lds_off_of(&As[row][8 * j]));
        }
        // B tile 32x64, transpose into Bs[n][k]
        #pragma unroll
        for (int i = 0; i < 8; ++i) {
            int idx = i * 128 + tid;
            int kk  = idx >> 5;
            int c2  = idx & 31;
            const __bf16* p2 = Bm + (size_t)(k0 + kk) * N + n0 + 2 * c2;
            Bs[2*c2    ][kk] = p2[0];
            Bs[2*c2 + 1][kk] = p2[1];
        }
        wait_async_0();
        __syncthreads();

        FragBF fa[2], fb[2];
        #pragma unroll
        for (int mi = 0; mi < 2; ++mi) {
            int row = wm + mi*16 + l16;
            #pragma unroll
            for (int p = 0; p < 8; ++p) {
                int k2 = kbB + 2*p;
                fa[mi].u[p] = *(const unsigned*)&As[row][k2];
            }
        }
        #pragma unroll
        for (int ni = 0; ni < 2; ++ni) {
            int col = wn + ni*16 + l16;
            #pragma unroll
            for (int p = 0; p < 8; ++p)
                fb[ni].u[p] = *(const unsigned*)&Bs[col][kbB + 2*p];
        }
        #pragma unroll
        for (int mi = 0; mi < 2; ++mi)
            #pragma unroll
            for (int ni = 0; ni < 2; ++ni)
                acc[mi][ni] = __builtin_amdgcn_wmma_f32_16x16x32_bf16(
                    false, fa[mi].v, false, fb[ni].v, (short)0, acc[mi][ni], false, false);
        __syncthreads();
    }

    #pragma unroll
    for (int mi = 0; mi < 2; ++mi) {
        #pragma unroll
        for (int ni = 0; ni < 2; ++ni) {
            int gn = n0 + wn + ni*16 + l16;
            float bv = bias[gn];
            #pragma unroll
            for (int r = 0; r < 8; ++r) {
                int gm = m0 + wm + mi*16 + halo*8 + r;
                float val = acc[mi][ni][r] + bv;
                int b = gm / SEQ;
                int s = gm - b * SEQ;
                if (mode == 0) {
                    int h = gn >> 6, d = gn & 63;
                    ((__bf16*)outp)[(((size_t)b*HEADS + h)*SEQ + s)*HDIM + d] = (__bf16)val;
                } else {
                    ((float*)outp)[((size_t)b*HID + gn)*SEQ + s] = val;
                }
            }
        }
    }
}

// ---------------- FlashAttention-style kernel (64 q-rows x head per block) ----------------
__global__ __launch_bounds__(128) void attention_wmma(
    const __bf16* __restrict__ Qp,   // [B,H,S,Dh]
    const __bf16* __restrict__ Kp,
    const __bf16* __restrict__ Vp,
    const int*    __restrict__ mask, // [B,S]
    __bf16* __restrict__ X)          // [B,S,HID] merged heads
{
    __shared__ __bf16 Ks[64][72];    // [key][d], 144B rows: async b128 + conflict-free frag reads
    __shared__ __bf16 Vs[64][72];    // [d][key] transposed on fill
    __shared__ __bf16 Ps[64][66];    // wave-private P staging [qrow][key]
    __shared__ int smask[64];

    const int tid  = threadIdx.x;
    const int lane = tid & 31;
    const int wave = tid >> 5;
    const int qt = blockIdx.x;
    const int bh = blockIdx.y;       // b*HEADS + h
    const int b  = bh >> 4;
    const int h  = bh & 15;

    const int halo = lane >> 4;
    const int l16  = lane & 15;
    const int kbA  = halo * 8;
    const int kbB  = halo * 16;

    const size_t head_base = (size_t)bh * SEQ * HDIM;

    // Q fragments resident in registers (A layout), 2 k-steps over d
    FragBF aq[2];
    {
        int qrow = qt*64 + wave*16 + l16;
        const __bf16* qptr = Qp + head_base + (size_t)qrow * HDIM;
        #pragma unroll
        for (int s2 = 0; s2 < 2; ++s2)
            #pragma unroll
            for (int p = 0; p < 8; ++p) {
                int d = s2*32 + kbB + 2*p;
                aq[s2].u[p] = *(const unsigned*)(qptr + d);
            }
    }

    v8f o[4] = {};
    float mrun[8], lrun[8];
    #pragma unroll
    for (int r = 0; r < 8; ++r) { mrun[r] = -1e30f; lrun[r] = 0.f; }

    for (int kt = 0; kt < SEQ/64; ++kt) {
        // K tile via async b128: 512 chunks of 16B, 4 per thread
        #pragma unroll
        for (int i = 0; i < 4; ++i) {
            int c   = i * 128 + tid;
            int row = c >> 3;            // 8 chunks per 64-elem row
            int j   = c & 7;
            async_copy_b128(Kp + head_base + (size_t)(kt*64 + row) * HDIM + 8 * j,
                            lds_off_of(&Ks[row][8 * j]));
        }
        // V tile transposed into Vs[d][key]
        #pragma unroll
        for (int i = 0; i < 16; ++i) {
            int idx = i*128 + tid;       // 2048 dword-pairs
            int key = idx >> 5;
            int c2  = idx & 31;
            const __bf16* vp2 = Vp + head_base + (size_t)(kt*64 + key)*HDIM + 2*c2;
            Vs[2*c2  ][key] = vp2[0];
            Vs[2*c2+1][key] = vp2[1];
        }
        if (tid < 64) smask[tid] = mask[(size_t)b*SEQ + kt*64 + tid];
        if (kt + 1 < SEQ/64)   // gfx1250 global_prefetch of next V tile
            __builtin_prefetch(Vp + head_base + (size_t)((kt+1)*64)*HDIM + tid*32, 0, 0);
        wait_async_0();
        __syncthreads();

        // scores: 16(q) x 64(key), K-dim = Dh = 64
        v8f sc[4];
        #pragma unroll
        for (int nt = 0; nt < 4; ++nt) {
            FragBF fb0, fb1;
            int col = nt*16 + l16;
            #pragma unroll
            for (int p = 0; p < 8; ++p) {
                fb0.u[p] = *(const unsigned*)&Ks[col][kbB + 2*p];
                fb1.u[p] = *(const unsigned*)&Ks[col][32 + kbB + 2*p];
            }
            v8f z = {};
            z = __builtin_amdgcn_wmma_f32_16x16x32_bf16(false, aq[0].v, false, fb0.v, (short)0, z, false, false);
            z = __builtin_amdgcn_wmma_f32_16x16x32_bf16(false, aq[1].v, false, fb1.v, (short)0, z, false, false);
            sc[nt] = z;
        }

        // scale 1/sqrt(64) + mask
        #pragma unroll
        for (int nt = 0; nt < 4; ++nt) {
            int msk = smask[nt*16 + l16];
            #pragma unroll
            for (int r = 0; r < 8; ++r) {
                float sv = sc[nt][r] * 0.125f;
                sc[nt][r] = (msk == 0) ? -1e9f : sv;
            }
        }

        // online softmax: row reductions across 16-lane half (C-layout N dim)
        float alpha[8];
        #pragma unroll
        for (int r = 0; r < 8; ++r) {
            float lm = fmaxf(fmaxf(sc[0][r], sc[1][r]), fmaxf(sc[2][r], sc[3][r]));
            #pragma unroll
            for (int mm = 1; mm < 16; mm <<= 1)
                lm = fmaxf(lm, __shfl_xor(lm, mm, 32));
            float mnew = fmaxf(mrun[r], lm);
            alpha[r] = __expf(mrun[r] - mnew);
            mrun[r] = mnew;
            float psum = 0.f;
            #pragma unroll
            for (int nt = 0; nt < 4; ++nt) {
                float p = __expf(sc[nt][r] - mnew);
                sc[nt][r] = p;
                psum += p;
            }
            #pragma unroll
            for (int mm = 1; mm < 16; mm <<= 1)
                psum += __shfl_xor(psum, mm, 32);
            lrun[r] = lrun[r] * alpha[r] + psum;
        }

        // rescale O; stage P (C-layout -> LDS, wave-private, same-wave DS ordering)
        #pragma unroll
        for (int nt = 0; nt < 4; ++nt)
            #pragma unroll
            for (int r = 0; r < 8; ++r) {
                o[nt][r] *= alpha[r];
                Ps[wave*16 + halo*8 + r][nt*16 + l16] = (__bf16)sc[nt][r];
            }

        // P @ V, K-dim = 64 keys
        FragBF ap[2];
        {
            int prow = wave*16 + l16;
            #pragma unroll
            for (int s2 = 0; s2 < 2; ++s2)
                #pragma unroll
                for (int p = 0; p < 8; ++p) {
                    int kk = s2*32 + kbB + 2*p;
                    ap[s2].u[p] = *(const unsigned*)&Ps[prow][kk];
                }
        }
        #pragma unroll
        for (int nt = 0; nt < 4; ++nt) {
            FragBF bv0, bv1;
            int dcol = nt*16 + l16;
            #pragma unroll
            for (int p = 0; p < 8; ++p) {
                bv0.u[p] = *(const unsigned*)&Vs[dcol][kbB + 2*p];
                bv1.u[p] = *(const unsigned*)&Vs[dcol][32 + kbB + 2*p];
            }
            o[nt] = __builtin_amdgcn_wmma_f32_16x16x32_bf16(false, ap[0].v, false, bv0.v, (short)0, o[nt], false, false);
            o[nt] = __builtin_amdgcn_wmma_f32_16x16x32_bf16(false, ap[1].v, false, bv1.v, (short)0, o[nt], false, false);
        }
        __syncthreads();
    }

    // normalize + merge heads to [B,S,HID] bf16
    #pragma unroll
    for (int nt = 0; nt < 4; ++nt)
        #pragma unroll
        for (int r = 0; r < 8; ++r) {
            int qrow = qt*64 + wave*16 + halo*8 + r;
            int d = nt*16 + l16;
            float val = o[nt][r] / lrun[r];
            X[((size_t)b*SEQ + qrow)*HID + h*HDIM + d] = (__bf16)val;
        }
}

extern "C" void kernel_launch(void* const* d_in, const int* in_sizes, int n_in,
                              void* d_out, int out_size, void* d_ws, size_t ws_size,
                              hipStream_t stream) {
    const float* q   = (const float*)d_in[0];
    const float* k   = (const float*)d_in[1];
    const float* v   = (const float*)d_in[2];
    const int*   msk = (const int*)d_in[3];
    const float* Wq  = (const float*)d_in[4];
    const float* bq  = (const float*)d_in[5];
    const float* Wk  = (const float*)d_in[6];
    const float* bk  = (const float*)d_in[7];
    const float* Wv  = (const float*)d_in[8];
    const float* bv  = (const float*)d_in[9];
    const float* Wo  = (const float*)d_in[10];
    const float* bo  = (const float*)d_in[11];

    const size_t NIN = (size_t)BATCH * SEQ * HID;  // 4 Mi elements
    const size_t NW  = (size_t)HID * HID;          // 1 Mi elements

    char* ws = (char*)d_ws;
    __bf16* qb  = (__bf16*)ws; ws += NIN*2;
    __bf16* kb  = (__bf16*)ws; ws += NIN*2;
    __bf16* vb  = (__bf16*)ws; ws += NIN*2;
    __bf16* wqb = (__bf16*)ws; ws += NW*2;
    __bf16* wkb = (__bf16*)ws; ws += NW*2;
    __bf16* wvb = (__bf16*)ws; ws += NW*2;
    __bf16* wob = (__bf16*)ws; ws += NW*2;
    __bf16* Qp  = (__bf16*)ws; ws += NIN*2;
    __bf16* Kp  = (__bf16*)ws; ws += NIN*2;
    __bf16* Vp  = (__bf16*)ws; ws += NIN*2;
    __bf16* X   = (__bf16*)ws; ws += NIN*2;

    const int thr = 256;
    cvt_f32_to_bf16<<<(int)((NIN + thr - 1) / thr), thr, 0, stream>>>(q, qb, (int)NIN);
    cvt_f32_to_bf16<<<(int)((NIN + thr - 1) / thr), thr, 0, stream>>>(k, kb, (int)NIN);
    cvt_f32_to_bf16<<<(int)((NIN + thr - 1) / thr), thr, 0, stream>>>(v, vb, (int)NIN);
    cvt_f32_to_bf16<<<(int)((NW  + thr - 1) / thr), thr, 0, stream>>>(Wq, wqb, (int)NW);
    cvt_f32_to_bf16<<<(int)((NW  + thr - 1) / thr), thr, 0, stream>>>(Wk, wkb, (int)NW);
    cvt_f32_to_bf16<<<(int)((NW  + thr - 1) / thr), thr, 0, stream>>>(Wv, wvb, (int)NW);
    cvt_f32_to_bf16<<<(int)((NW  + thr - 1) / thr), thr, 0, stream>>>(Wo, wob, (int)NW);

    dim3 gg(BATCH * SEQ / 64, HID / 64);
    gemm_bf16_wmma<<<gg, 128, 0, stream>>>(qb, wqb, bq, Qp, BATCH*SEQ, HID, HID, 0);
    gemm_bf16_wmma<<<gg, 128, 0, stream>>>(kb, wkb, bk, Kp, BATCH*SEQ, HID, HID, 0);
    gemm_bf16_wmma<<<gg, 128, 0, stream>>>(vb, wvb, bv, Vp, BATCH*SEQ, HID, HID, 0);

    attention_wmma<<<dim3(SEQ/64, BATCH*HEADS), 128, 0, stream>>>(Qp, Kp, Vp, msk, X);

    gemm_bf16_wmma<<<gg, 128, 0, stream>>>(X, wob, bo, d_out, BATCH*SEQ, HID, HID, 1);
}
